// MoEDispatchCombine_32306744000740
// MI455X (gfx1250) — matrix operand order repaired and checked by reading, not compiled, not run on hardware
//
#include <hip/hip_runtime.h>
#include <hip/hip_bf16.h>

typedef __attribute__((ext_vector_type(16))) __bf16 v16bf;
typedef __attribute__((ext_vector_type(8)))  __bf16 v8bf;
typedef __attribute__((ext_vector_type(4)))  __bf16 v4bf;
typedef __attribute__((ext_vector_type(8)))  float  v8f;

// ---- problem constants (A = 32) ----
#define NNODE   8192
#define NEDGE   65536
#define NANGLE  131072
#define D_ND    128      // nd
#define D_NSYM  768      // n_sym
#define D_EINFO 320      // edge_info
#define D_ADIM  128      // angle dim
#define DOUT_E  192      // nd + ne
#define DOUT_A  96       // ne + na
#define EXPN    8
#define TOPK    2

// padded per-table assignment capacities (TOPK*M + 16 pad per expert)
#define CAPN (TOPK*NNODE  + 16*EXPN)   // 16512
#define CAPE (TOPK*NEDGE  + 16*EXPN)   // 131200
#define CAPA (TOPK*NANGLE + 16*EXPN)   // 262272

// GEMM tile grids
#define TILESN (CAPN/16)   // 1032
#define TILESE (CAPE/16)   // 8200
#define TILESA (CAPA/16)   // 16392

// packed-B element counts: EXPN * (din/32) * (dout/16) * 512
#define NW1P (EXPN*(D_ND  /32)*(D_ND  /16)*512)  // 131072
#define NW2P (EXPN*(D_NSYM/32)*(D_ND  /16)*512)  // 786432
#define NWEP (EXPN*(D_EINFO/32)*(DOUT_E/16)*512) // 491520
#define NWAP (EXPN*(D_ADIM/32)*(DOUT_A/16)*512)  // 98304

// workspace byte offsets (each region 256B-aligned)
#define ALIGN256(x) (((x) + 255) & ~(size_t)255)
#define OFF_CTR   ((size_t)0)                                   // 96 ints
#define OFF_RIDN  ALIGN256(OFF_CTR  + 96*4)
#define OFF_WGTN  ALIGN256(OFF_RIDN + (size_t)CAPN*4)
#define OFF_RIDE  ALIGN256(OFF_WGTN + (size_t)CAPN*4)
#define OFF_WGTE  ALIGN256(OFF_RIDE + (size_t)CAPE*4)
#define OFF_RIDA  ALIGN256(OFF_WGTE + (size_t)CAPE*4)
#define OFF_WGTA  ALIGN256(OFF_RIDA + (size_t)CAPA*4)
#define OFF_W1P   ALIGN256(OFF_WGTA + (size_t)CAPA*4)
#define OFF_W2P   ALIGN256(OFF_W1P  + (size_t)NW1P*2)
#define OFF_WEP   ALIGN256(OFF_W2P  + (size_t)NW2P*2)
#define OFF_WAP   ALIGN256(OFF_WEP  + (size_t)NWEP*2)

// ctrs layout (ints): [0..23]=cnt[3][8], [24..47]=segStart, [48..71]=segEnd, [72..95]=fill

// ---------------- init: zero output, counters; rid = -1, wgt = 0 ----------------
__global__ void k_init(float* __restrict__ out, int nout,
                       int* __restrict__ ridN, int* __restrict__ ridE, int* __restrict__ ridA,
                       float* __restrict__ wN, float* __restrict__ wE, float* __restrict__ wA,
                       int* __restrict__ ctrs)
{
    int i = blockIdx.x * blockDim.x + threadIdx.x;
    if (i < nout) out[i] = 0.0f;
    if (i < CAPN) { ridN[i] = -1; wN[i] = 0.0f; }
    if (i < CAPE) { ridE[i] = -1; wE[i] = 0.0f; }
    if (i < CAPA) { ridA[i] = -1; wA[i] = 0.0f; }
    if (i < 96)   ctrs[i] = 0;
}

// ---------------- routing: count, offsets, scatter ----------------
__global__ void k_count(const int* __restrict__ idxTab, const int* __restrict__ map,
                        int n, int* __restrict__ cnt)
{
    int i = blockIdx.x * blockDim.x + threadIdx.x;
    if (i >= n) return;
    int r = map ? map[i] : i;
    atomicAdd(&cnt[idxTab[2*r + 0]], 1);
    atomicAdd(&cnt[idxTab[2*r + 1]], 1);
}

__global__ void k_offsets(int* __restrict__ ctrs)
{
    if (blockIdx.x != 0 || threadIdx.x != 0) return;
    for (int t = 0; t < 3; ++t) {
        int s = 0;
        for (int e = 0; e < EXPN; ++e) {
            int c = ctrs[t*8 + e];
            ctrs[24 + t*8 + e] = s;                 // segStart (16-aligned)
            int pe = s + ((c + 15) & ~15);
            ctrs[48 + t*8 + e] = pe;                // segEnd (padded)
            ctrs[72 + t*8 + e] = 0;                 // fill
            s = pe;
        }
    }
}

__global__ void k_scatter(const int* __restrict__ idxTab, const float* __restrict__ wTab,
                          const int* __restrict__ map, int n,
                          const int* __restrict__ ss, int* __restrict__ fill,
                          int* __restrict__ rid, float* __restrict__ wgt)
{
    int i = blockIdx.x * blockDim.x + threadIdx.x;
    if (i >= n) return;
    int r = map ? map[i] : i;
    #pragma unroll
    for (int k = 0; k < TOPK; ++k) {
        int e = idxTab[2*r + k];
        int pos = ss[e] + atomicAdd(&fill[e], 1);
        rid[pos] = i;
        wgt[pos] = wTab[2*r + k];
    }
}

// ---------------- pack W (f32 [E,din,dout]) into bf16 B-fragment order ----------------
// per (e,kt,nt) tile of 512 bf16: lane l owns 16 elems; B layout (ISA 7.12.2):
// col = l&15, K = kt*32 + (l>>4)*8 + (j&7) + (j>>3)*16
__global__ void k_packB(const float* __restrict__ W, __bf16* __restrict__ out,
                        int din, int dout)
{
    int KT = din >> 5, NT = dout >> 4;
    int total = EXPN * KT * NT * 512;
    int t = blockIdx.x * blockDim.x + threadIdx.x;
    if (t >= total) return;
    int j   = t & 15;
    int l   = (t >> 4) & 31;
    int rem = t >> 9;
    int nt  = rem % NT; rem /= NT;
    int kt  = rem % KT;
    int e   = rem / KT;
    int col  = nt*16 + (l & 15);
    int krow = kt*32 + (l >> 4)*8 + (j & 7) + (j >> 3)*16;
    out[t] = (__bf16)W[((size_t)e * din + krow) * dout + col];
}

// ---------------- expert-tiled GEMM + SiLU + gate + atomic combine ----------------
// One block = one 16-row single-expert tile. NWAVES waves, NT/NWAVES accumulators
// per wave: A fragment loaded once per k-step and reused across N-tiles.
template<int DIN, int DOUT, int NWAVES>
__global__ __launch_bounds__(NWAVES * 32)
void k_gemm(const float* __restrict__ X, const __bf16* __restrict__ Wp,
            const float* __restrict__ bias, const float* __restrict__ wgt,
            const int* __restrict__ rid, const int* __restrict__ ss,
            const int* __restrict__ se, float* __restrict__ out)
{
    constexpr int KT  = DIN  / 32;
    constexpr int NT  = DOUT / 16;
    constexpr int NTP = NT / NWAVES;            // accumulators per wave
    constexpr int BD  = NWAVES * 32;
    static_assert(NTP * NWAVES == NT, "NT must divide evenly");

    const int base = blockIdx.x * 16;
    int expert = -1;
    #pragma unroll
    for (int e = 0; e < EXPN; ++e)
        if (expert < 0 && base >= ss[e] && base < se[e]) expert = e;
    if (expert < 0) return;   // beyond padded total (block-uniform)

    __shared__ __bf16 xs[16 * DIN];
    __shared__ int    srid[16];
    __shared__ float  swgt[16];

    const int tid = threadIdx.x;
    if (tid < 16) { srid[tid] = rid[base + tid]; swgt[tid] = wgt[base + tid]; }
    __syncthreads();

    // gather + convert 16 activation rows into LDS via 128-bit loads
    constexpr int C4 = DIN / 4;                 // float4 per row
    for (int idx = tid; idx < 16 * C4; idx += BD) {
        int rr = idx / C4, c4 = idx - rr * C4;
        int rg = srid[rr];
        float4 f = make_float4(0.f, 0.f, 0.f, 0.f);
        if (rg >= 0) f = *(const float4*)(X + (size_t)rg * DIN + c4 * 4);
        v4bf h = { (__bf16)f.x, (__bf16)f.y, (__bf16)f.z, (__bf16)f.w };
        *(v4bf*)(xs + rr * DIN + c4 * 4) = h;   // ds_store_b64
    }
    __syncthreads();

    const int lane = tid & 31;
    const int wave = tid >> 5;
    const int arow = lane & 15;           // A-matrix M row for this lane
    const int kb   = (lane >> 4) * 8;     // A-fragment K sub-base

    v8f acc[NTP];
    #pragma unroll
    for (int i = 0; i < NTP; ++i)
        acc[i] = (v8f){0.f, 0.f, 0.f, 0.f, 0.f, 0.f, 0.f, 0.f};

    for (int kt = 0; kt < KT; ++kt) {
        // A fragment (shared by all N-tiles of this wave):
        // elems 0..7 -> K=kb..kb+7, elems 8..15 -> K=kb+16..kb+23
        const __bf16* ap = xs + arow * DIN + kt * 32 + kb;
        v8bf a0 = *(const v8bf*)ap;
        v8bf a1 = *(const v8bf*)(ap + 16);
        v16bf a;
        #pragma unroll
        for (int j = 0; j < 8; ++j) { a[j] = a0[j]; a[8 + j] = a1[j]; }

        #pragma unroll
        for (int i = 0; i < NTP; ++i) {
            const int nt = wave + i * NWAVES;
            // B fragment: pre-packed, lane-contiguous 16 bf16 (32B)
            const __bf16* bp = Wp + ((((size_t)expert * KT + kt) * NT + nt) * 32 + lane) * 16;
            v8bf b0 = *(const v8bf*)bp;
            v8bf b1 = *(const v8bf*)(bp + 8);
            v16bf b;
            #pragma unroll
            for (int j = 0; j < 8; ++j) { b[j] = b0[j]; b[8 + j] = b1[j]; }

            acc[i] = __builtin_amdgcn_wmma_f32_16x16x32_bf16(
                         false, a, false, b, (short)0, acc[i], false, false);
        }
    }

    // epilogue: C layout VGPR v -> row v + 8*(lane>=16); col = lane&15
    const int rsel = (lane >> 4) * 8;
    #pragma unroll
    for (int i = 0; i < NTP; ++i) {
        const int nt  = wave + i * NWAVES;
        const int col = nt * 16 + (lane & 15);
        const float bv = bias[expert * DOUT + col];
        #pragma unroll
        for (int v = 0; v < 8; ++v) {
            int rloc = v + rsel;
            int rg = srid[rloc];
            if (rg >= 0) {
                float z = acc[i][v] + bv;
                // SiLU via fast reciprocal (v_exp_f32 + v_rcp_f32), bf16 WMMA
                // already bounds precision; avoid IEEE div expansion.
                float sig = __builtin_amdgcn_rcpf(1.0f + __expf(-z));
                atomicAdd(out + (size_t)rg * DOUT + col, z * sig * swgt[rloc]);
            }
        }
    }
}

// ---------------- launch ----------------
extern "C" void kernel_launch(void* const* d_in, const int* in_sizes, int n_in,
                              void* d_out, int out_size, void* d_ws, size_t ws_size,
                              hipStream_t stream)
{
    const float* x_m1   = (const float*)d_in[0];
    const float* x_m2   = (const float*)d_in[1];
    const float* x_ed   = (const float*)d_in[2];
    const float* x_an   = (const float*)d_in[3];
    const float* nw     = (const float*)d_in[4];
    const int*   ni     = (const int*)  d_in[5];
    const float* ewn    = (const float*)d_in[6];
    const int*   ein    = (const int*)  d_in[7];
    const float* awn    = (const float*)d_in[8];
    const int*   ain    = (const int*)  d_in[9];
    const int*   n2e    = (const int*)  d_in[10];
    const int*   n2a    = (const int*)  d_in[11];
    const float* W1 = (const float*)d_in[12]; const float* b1 = (const float*)d_in[13];
    const float* W2 = (const float*)d_in[14]; const float* b2 = (const float*)d_in[15];
    const float* We = (const float*)d_in[16]; const float* be = (const float*)d_in[17];
    const float* Wa = (const float*)d_in[18]; const float* ba = (const float*)d_in[19];

    char* ws = (char*)d_ws;
    int*    ctrs = (int*)  (ws + OFF_CTR);
    int*    ridN = (int*)  (ws + OFF_RIDN);  float* wgtN = (float*)(ws + OFF_WGTN);
    int*    ridE = (int*)  (ws + OFF_RIDE);  float* wgtE = (float*)(ws + OFF_WGTE);
    int*    ridA = (int*)  (ws + OFF_RIDA);  float* wgtA = (float*)(ws + OFF_WGTA);
    __bf16* W1p  = (__bf16*)(ws + OFF_W1P);
    __bf16* W2p  = (__bf16*)(ws + OFF_W2P);
    __bf16* Wep  = (__bf16*)(ws + OFF_WEP);
    __bf16* Wap  = (__bf16*)(ws + OFF_WAP);

    float* out  = (float*)d_out;
    float* oM1  = out;
    float* oM2  = out + (size_t)NNODE * D_ND;
    float* oED  = oM2 + (size_t)NNODE * D_ND;
    float* oAN  = oED + (size_t)NEDGE * DOUT_E;

    // counters index helpers
    int* cntN = ctrs + 0;  int* cntE = ctrs + 8;  int* cntA = ctrs + 16;
    int* ssN  = ctrs + 24; int* ssE  = ctrs + 32; int* ssA  = ctrs + 40;
    int* seN  = ctrs + 48; int* seE  = ctrs + 56; int* seA  = ctrs + 64;
    int* flN  = ctrs + 72; int* flE  = ctrs + 80; int* flA  = ctrs + 88;

    // 1) init
    {
        int n = out_size > CAPA ? out_size : CAPA;
        k_init<<<(n + 255) / 256, 256, 0, stream>>>(out, out_size,
            ridN, ridE, ridA, wgtN, wgtE, wgtA, ctrs);
    }
    // 2) counts
    k_count<<<(NNODE  + 255) / 256, 256, 0, stream>>>(ni,  nullptr, NNODE,  cntN);
    k_count<<<(NEDGE  + 255) / 256, 256, 0, stream>>>(ein, n2e,     NEDGE,  cntE);
    k_count<<<(NANGLE + 255) / 256, 256, 0, stream>>>(ain, n2a,     NANGLE, cntA);
    // 3) segment offsets
    k_offsets<<<1, 1, 0, stream>>>(ctrs);
    // 4) scatter
    k_scatter<<<(NNODE  + 255) / 256, 256, 0, stream>>>(ni,  nw,  nullptr, NNODE,  ssN, flN, ridN, wgtN);
    k_scatter<<<(NEDGE  + 255) / 256, 256, 0, stream>>>(ein, ewn, n2e,     NEDGE,  ssE, flE, ridE, wgtE);
    k_scatter<<<(NANGLE + 255) / 256, 256, 0, stream>>>(ain, awn, n2a,     NANGLE, ssA, flA, ridA, wgtA);
    // 5) pack weights to bf16 fragment order
    k_packB<<<(NW1P + 255) / 256, 256, 0, stream>>>(W1, W1p, D_ND,    D_ND);
    k_packB<<<(NW2P + 255) / 256, 256, 0, stream>>>(W2, W2p, D_NSYM,  D_ND);
    k_packB<<<(NWEP + 255) / 256, 256, 0, stream>>>(We, Wep, D_EINFO, DOUT_E);
    k_packB<<<(NWAP + 255) / 256, 256, 0, stream>>>(Wa, Wap, D_ADIM,  DOUT_A);
    // 6) expert GEMMs (templated: NT/NWAVES accumulators per wave)
    k_gemm<D_ND,    D_ND,   4><<<TILESN, 128, 0, stream>>>(x_m1, W1p, b1, wgtN, ridN, ssN, seN, oM1);
    k_gemm<D_NSYM,  D_ND,   4><<<TILESN, 128, 0, stream>>>(x_m2, W2p, b2, wgtN, ridN, ssN, seN, oM2);
    k_gemm<D_EINFO, DOUT_E, 4><<<TILESE, 128, 0, stream>>>(x_ed, Wep, be, wgtE, ridE, ssE, seE, oED);
    k_gemm<D_ADIM,  DOUT_A, 3><<<TILESA,  96, 0, stream>>>(x_an, Wap, ba, wgtA, ridA, ssA, seA, oAN);
}
